// DecoderTopDown_75402445848592
// MI455X (gfx1250) — matrix operand order repaired and checked by reading, not compiled, not run on hardware
//
#include <hip/hip_runtime.h>
#include <hip/hip_bf16.h>
#include <math.h>

// ---------------------------------------------------------------------------
// Beam-search LSTM decoder for MI455X (gfx1250, wave32, WMMA).
// GEMMs on v_wmma_f32_16x16x32_bf16, one wave per 16-wide N tile with 6
// M-tile accumulators. Weights pre-packed once per launch into the bf16
// B-operand tile layout (halves L2 weight traffic vs fp32 streaming and
// removes per-iteration cvt); software-pipelined inner loop (double-buffered
// A/B registers + global_prefetch on the weight stream). Falls back to
// fp32-streaming weights if the workspace is too small for packed weights.
// ---------------------------------------------------------------------------

typedef __bf16 bf16;
typedef bf16  v16bf __attribute__((ext_vector_type(16)));
typedef float v8f   __attribute__((ext_vector_type(8)));

#define CB 32
#define CK 3
#define CT 20
#define CV 10000
#define CR 36
#define C_E 512
#define C_VD 2048
#define C_H 1024
#define C_PH 256
#define ROWS 96           // B*K
#define NEGV (-1e9f)

__device__ __forceinline__ float sigf(float x) { return 1.0f / (1.0f + expf(-x)); }

// A-operand element->K mapping for 16x16x32 bf16 WMMA:
// lane L: M = L%16, half = L/16; elems 0..15 -> K = half*16 + e
// (contiguous, matching the B packing).
__device__ __forceinline__ int a_k_of(int kt, int half, int e) {
  return kt * 32 + half * 16 + e;
}

// ---------------------------------------------------------------------------
// Packed-weight WMMA GEMM: C[M,Nreal] = Apacked * Bpacked (+bias).
// Both operands pre-packed in per-lane tile layout; inner loop double-buffers
// next k-tile's A/B registers so loads overlap the WMMAs.
// ---------------------------------------------------------------------------
template <int MT>
__global__ __launch_bounds__(32) void gemm_pk_wmma(
    const bf16* __restrict__ Ap, const bf16* __restrict__ Bp, int Ktiles,
    const float* __restrict__ bias, float* __restrict__ C, int Nreal) {
  const int lane = threadIdx.x;
  const int nt = blockIdx.x;
  const int mbase = blockIdx.y * MT;
  const int n = nt * 16 + (lane & 15);

  const bf16* bptr = Bp + ((size_t)nt * Ktiles * 32 + lane) * 16;

  v8f z;
#pragma unroll
  for (int r = 0; r < 8; ++r) z[r] = 0.0f;
  v8f acc[MT];
#pragma unroll
  for (int i = 0; i < MT; ++i) acc[i] = z;

  v16bf aC[MT], bC;
  bC = *(const v16bf*)(bptr);
#pragma unroll
  for (int i = 0; i < MT; ++i)
    aC[i] = *(const v16bf*)(Ap + ((size_t)(mbase + i) * Ktiles * 32 + lane) * 16);

  for (int kt = 0; kt < Ktiles; ++kt) {
    const int ktn = (kt + 1 < Ktiles) ? kt + 1 : kt;   // clamped redundant tail load
    v16bf aN[MT], bN;
    bN = *(const v16bf*)(bptr + (size_t)ktn * 512);
#pragma unroll
    for (int i = 0; i < MT; ++i)
      aN[i] = *(const v16bf*)(Ap + ((size_t)((mbase + i) * Ktiles + ktn) * 32 + lane) * 16);
    if (kt + 8 < Ktiles)
      __builtin_prefetch(bptr + (size_t)(kt + 8) * 512, 0, 0);
#pragma unroll
    for (int i = 0; i < MT; ++i)
      acc[i] = __builtin_amdgcn_wmma_f32_16x16x32_bf16(
          false, aC[i], false, bC, (short)0, acc[i], false, false);
    bC = bN;
#pragma unroll
    for (int i = 0; i < MT; ++i) aC[i] = aN[i];
  }

  if (n < Nreal) {
    const float bsum = bias ? bias[n] : 0.0f;
#pragma unroll
    for (int i = 0; i < MT; ++i) {
      const int mrow = (mbase + i) * 16 + (lane >> 4) * 8;
#pragma unroll
      for (int r = 0; r < 8; ++r)
        C[(size_t)(mrow + r) * Nreal + n] = acc[i][r] + bsum;
    }
  }
}

// ---------------------------------------------------------------------------
// Fallback WMMA GEMM streaming fp32 weights [W0 | W1] with in-register cvt.
// ---------------------------------------------------------------------------
template <int MT>
__global__ __launch_bounds__(32) void gemm_bf16_wmma(
    const bf16* __restrict__ Ap, int Ktiles,
    const float* __restrict__ W0, int K0,
    const float* __restrict__ W1, int K1,
    const float* __restrict__ bias,
    float* __restrict__ C, int Nreal) {
  const int lane = threadIdx.x;
  const int nt = blockIdx.x;
  const int mbase = blockIdx.y * MT;
  const int n = nt * 16 + (lane & 15);
  const int khalf = (lane >> 4) * 16;
  const bool nok = (n < Nreal);

  v8f z;
#pragma unroll
  for (int r = 0; r < 8; ++r) z[r] = 0.0f;
  v8f acc[MT];
#pragma unroll
  for (int i = 0; i < MT; ++i) acc[i] = z;

  for (int kt = 0; kt < Ktiles; ++kt) {
    const int k0 = kt * 32 + khalf;
    v16bf bv;
    if (nok) {
      const float* wrow;
      if (k0 < K0) wrow = W0 + (size_t)n * K0 + k0;
      else         wrow = W1 + (size_t)n * K1 + (k0 - K0);
      const float4* wp = (const float4*)wrow;
      float4 w0 = wp[0], w1 = wp[1], w2 = wp[2], w3 = wp[3];
      bv[0] = (bf16)w0.x; bv[1] = (bf16)w0.y; bv[2] = (bf16)w0.z; bv[3] = (bf16)w0.w;
      bv[4] = (bf16)w1.x; bv[5] = (bf16)w1.y; bv[6] = (bf16)w1.z; bv[7] = (bf16)w1.w;
      bv[8] = (bf16)w2.x; bv[9] = (bf16)w2.y; bv[10] = (bf16)w2.z; bv[11] = (bf16)w2.w;
      bv[12] = (bf16)w3.x; bv[13] = (bf16)w3.y; bv[14] = (bf16)w3.z; bv[15] = (bf16)w3.w;
    } else {
#pragma unroll
      for (int e = 0; e < 16; ++e) bv[e] = (bf16)0.0f;
    }
#pragma unroll
    for (int i = 0; i < MT; ++i) {
      const v16bf a = *(const v16bf*)(Ap + ((size_t)((mbase + i) * Ktiles + kt) * 32 + lane) * 16);
      acc[i] = __builtin_amdgcn_wmma_f32_16x16x32_bf16(
          false, a, false, bv, (short)0, acc[i], false, false);
    }
  }

  if (nok) {
    const float bsum = bias ? bias[n] : 0.0f;
#pragma unroll
    for (int i = 0; i < MT; ++i) {
      const int mrow = (mbase + i) * 16 + (lane >> 4) * 8;
#pragma unroll
      for (int r = 0; r < 8; ++r)
        C[(size_t)(mrow + r) * Nreal + n] = acc[i][r] + bsum;
    }
  }
}

// fp32 [Nreal, K0+K1] (two row-major segments) -> packed bf16 B-operand tiles.
// B layout: lane L holds N = nt*16 + L%16, K = kt*32 + (L/16)*16 + e.
__global__ void pack_b_kernel(const float* __restrict__ W0, int K0,
                              const float* __restrict__ W1, int K1,
                              bf16* __restrict__ dst, int Ntiles, int Ktiles, int Nreal) {
  const int idx = blockIdx.x * blockDim.x + threadIdx.x;
  const int total = Ntiles * Ktiles * 32;
  if (idx >= total) return;
  const int lane = idx & 31;
  const int kt = (idx >> 5) % Ktiles;
  const int nt = idx / (32 * Ktiles);
  const int n = nt * 16 + (lane & 15);
  const int k0 = kt * 32 + (lane >> 4) * 16;
  bf16* out = dst + (size_t)idx * 16;
  if (n >= Nreal) {
#pragma unroll
    for (int e = 0; e < 16; ++e) out[e] = (bf16)0.0f;
    return;
  }
  const float* wrow = (k0 < K0) ? (W0 + (size_t)n * K0 + k0)
                                : (W1 + (size_t)n * K1 + (k0 - K0));
#pragma unroll
  for (int e = 0; e < 16; ++e) out[e] = (bf16)wrow[e];
}

// Generic fp32 [Mtiles*16, Kd] row-major -> packed bf16 A-operand tiles.
__global__ void pack_a_kernel(const float* __restrict__ src, bf16* __restrict__ dst,
                              int Mtiles, int Ktiles, int Kd) {
  const int idx = blockIdx.x * blockDim.x + threadIdx.x;
  const int total = Mtiles * Ktiles * 32;
  if (idx >= total) return;
  const int lane = idx & 31;
  const int kt = (idx >> 5) % Ktiles;
  const int mt = idx / (32 * Ktiles);
  const int m = mt * 16 + (lane & 15);
  const int half = lane >> 4;
  bf16* out = dst + (size_t)idx * 16;
#pragma unroll
  for (int e = 0; e < 16; ++e) {
    const int k = a_k_of(kt, half, e);
    out[e] = (bf16)src[(size_t)m * Kd + k];
  }
}

// xcat1 = [h2(1024), vmean(2048), E[tok](512), h1(1024)]  -> packed A, K=4608
__global__ void concat1_pack(const int* __restrict__ tokens,
                             const float* __restrict__ h1c, const float* __restrict__ h2c,
                             const float* __restrict__ vmean, const float* __restrict__ E,
                             bf16* __restrict__ dst) {
  const int idx = blockIdx.x * blockDim.x + threadIdx.x;
  if (idx >= 6 * 144 * 32) return;
  const int lane = idx & 31;
  const int kt = (idx >> 5) % 144;
  const int mt = idx / (32 * 144);
  const int m = mt * 16 + (lane & 15);
  const int b = m / CK;
  const int tok = tokens[m];
  const int half = lane >> 4;
  bf16* out = dst + (size_t)idx * 16;
#pragma unroll
  for (int e = 0; e < 16; ++e) {
    const int k = a_k_of(kt, half, e);
    float v;
    if (k < 1024)      v = h2c[m * 1024 + k];
    else if (k < 3072) v = vmean[b * 2048 + (k - 1024)];
    else if (k < 3584) v = E[(size_t)tok * 512 + (k - 3072)];
    else               v = h1c[m * 1024 + (k - 3584)];
    out[e] = (bf16)v;
  }
}

// xcat2 = [h1(1024), vhat(2048), h2prev(1024)] -> packed A, K=4096
__global__ void concat2_pack(const float* __restrict__ h1t, const float* __restrict__ vhat,
                             const float* __restrict__ h2c, bf16* __restrict__ dst) {
  const int idx = blockIdx.x * blockDim.x + threadIdx.x;
  if (idx >= 6 * 128 * 32) return;
  const int lane = idx & 31;
  const int kt = (idx >> 5) % 128;
  const int mt = idx / (32 * 128);
  const int m = mt * 16 + (lane & 15);
  const int half = lane >> 4;
  bf16* out = dst + (size_t)idx * 16;
#pragma unroll
  for (int e = 0; e < 16; ++e) {
    const int k = a_k_of(kt, half, e);
    float v;
    if (k < 1024)      v = h1t[m * 1024 + k];
    else if (k < 3072) v = vhat[(size_t)m * 2048 + (k - 1024)];
    else               v = h2c[m * 1024 + (k - 3072)];
    out[e] = (bf16)v;
  }
}

// LSTM elementwise (torch gate order i,f,g,o in N quarters of the 4096 gates)
__global__ void lstm_ew(const float* __restrict__ gates, const float* __restrict__ cprev,
                        float* __restrict__ hout, float* __restrict__ cout) {
  const int idx = blockIdx.x * blockDim.x + threadIdx.x;
  if (idx >= ROWS * C_H) return;
  const int m = idx >> 10, j = idx & 1023;
  const float* g = gates + (size_t)m * 4096;
  const float gi = g[j], gf = g[1024 + j], gg = g[2048 + j], go = g[3072 + j];
  const float c = sigf(gf) * cprev[idx] + sigf(gi) * tanhf(gg);
  hout[idx] = sigf(go) * tanhf(c);
  cout[idx] = c;
}

// Additive attention: alpha = softmax_r( wa . tanh(featP[b,r,:] + hp[row,:]) ),
// vhat[row,:] = sum_r alpha_r * features[b,r,:]. Wave-parallel over r.
__global__ __launch_bounds__(256) void attention_kernel(
    const float* __restrict__ featP, const float* __restrict__ hp,
    const float* __restrict__ wa, const float* __restrict__ features,
    float* __restrict__ vhat) {
  const int row = blockIdx.x;
  const int b = row / CK;
  const int tid = threadIdx.x;
  const int wv = tid >> 5, ln = tid & 31;
  __shared__ float s_alpha[CR];
  for (int r = wv; r < CR; r += 8) {
    float acc = 0.0f;
    for (int p = ln; p < C_PH; p += 32)
      acc += tanhf(featP[((size_t)b * CR + r) * C_PH + p] + hp[row * C_PH + p]) * wa[p];
#pragma unroll
    for (int off = 16; off > 0; off >>= 1) acc += __shfl_down(acc, off);
    if (ln == 0) s_alpha[r] = acc;
  }
  __syncthreads();
  if (tid == 0) {
    float m = -3e38f;
    for (int r = 0; r < CR; ++r) m = fmaxf(m, s_alpha[r]);
    float ssum = 0.0f;
    for (int r = 0; r < CR; ++r) { s_alpha[r] = expf(s_alpha[r] - m); ssum += s_alpha[r]; }
    const float inv = 1.0f / ssum;
    for (int r = 0; r < CR; ++r) s_alpha[r] *= inv;
  }
  __syncthreads();
  for (int d = tid; d < C_VD; d += 256) {
    float acc = 0.0f;
    for (int r = 0; r < CR; ++r)
      acc += s_alpha[r] * features[((size_t)b * CR + r) * C_VD + d];
    vhat[(size_t)row * C_VD + d] = acc;
  }
}

// log_softmax + no-dup + EOS-row + trajectory score + t==0 beam mask
__global__ __launch_bounds__(256) void score_kernel(
    const float* __restrict__ logits, const int* __restrict__ tokens,
    const int* __restrict__ finished, const float* __restrict__ logp,
    int t, float* __restrict__ scores) {
  const int row = blockIdx.x;
  const int kb = row % CK;
  const int tid = threadIdx.x;
  __shared__ float sred[256];
  const float* L = logits + (size_t)row * CV;
  float m = -3e38f;
  for (int v = tid; v < CV; v += 256) m = fmaxf(m, L[v]);
  sred[tid] = m;
  __syncthreads();
  for (int s = 128; s > 0; s >>= 1) {
    if (tid < s) sred[tid] = fmaxf(sred[tid], sred[tid + s]);
    __syncthreads();
  }
  const float rowmax = sred[0];
  __syncthreads();
  float ssum = 0.0f;
  for (int v = tid; v < CV; v += 256) ssum += expf(L[v] - rowmax);
  sred[tid] = ssum;
  __syncthreads();
  for (int s = 128; s > 0; s >>= 1) {
    if (tid < s) sred[tid] += sred[tid + s];
    __syncthreads();
  }
  const float lse = rowmax + logf(sred[0]);
  const int ptok = tokens[row];
  const int fin = finished[row];
  const float lt = logp[row];
  float* S = scores + (size_t)row * 10016;
  for (int v = tid; v < 10016; v += 256) {
    float sc;
    if (v >= CV) {
      sc = NEGV;
    } else {
      float lp = L[v] - lse;
      if (t > 0 && v == ptok) lp -= 1e4f;
      if (fin) lp = (v == 2) ? 0.0f : NEGV;
      sc = lt + lp;
      if (t == 0 && kb > 0) sc = NEGV;
    }
    S[v] = sc;
  }
}

// top-3 over K*Vpad = 30048 per batch element
__global__ __launch_bounds__(256) void topk_kernel(
    const float* __restrict__ scores, int* __restrict__ tokens,
    float* __restrict__ logp, int* __restrict__ parent) {
  const int b = blockIdx.x;
  const int tid = threadIdx.x;
  const float* S = scores + (size_t)b * 3 * 10016;
  float v0 = -3.0e38f, v1 = -3.0e38f, v2 = -3.0e38f;
  int i0 = 0, i1 = 0, i2 = 0;
  for (int i = tid; i < 3 * 10016; i += 256) {
    const float v = S[i];
    if (v > v0)      { v2 = v1; i2 = i1; v1 = v0; i1 = i0; v0 = v; i0 = i; }
    else if (v > v1) { v2 = v1; i2 = i1; v1 = v;  i1 = i; }
    else if (v > v2) { v2 = v;  i2 = i; }
  }
  __shared__ float sv[768];
  __shared__ int si[768];
  sv[tid * 3 + 0] = v0; si[tid * 3 + 0] = i0;
  sv[tid * 3 + 1] = v1; si[tid * 3 + 1] = i1;
  sv[tid * 3 + 2] = v2; si[tid * 3 + 2] = i2;
  __syncthreads();
  if (tid == 0) {
    for (int r = 0; r < 3; ++r) {
      float m = -3.1e38f;
      int mslot = 0;
      for (int j = 0; j < 768; ++j)
        if (sv[j] > m) { m = sv[j]; mslot = j; }
      const int gi = si[mslot];
      sv[mslot] = -3.2e38f;
      parent[b * 3 + r] = gi / 10016;
      tokens[b * 3 + r] = gi % 10016;
      logp[b * 3 + r] = m;
    }
  }
}

__global__ void gather_states(const int* __restrict__ parent,
                              const float* __restrict__ h1t, float* __restrict__ h1c,
                              const float* __restrict__ c1t, float* __restrict__ c1c,
                              const float* __restrict__ h2t, float* __restrict__ h2c,
                              const float* __restrict__ c2t, float* __restrict__ c2c) {
  const int idx = blockIdx.x * blockDim.x + threadIdx.x;
  if (idx >= ROWS * C_H) return;
  const int m = idx >> 10, j = idx & 1023;
  const int b = m / CK;
  const int src = ((b * CK + parent[m]) << 10) | j;
  h1c[idx] = h1t[src];
  c1c[idx] = c1t[src];
  h2c[idx] = h2t[src];
  c2c[idx] = c2t[src];
}

__global__ void beam_update(const int* __restrict__ parent, const int* __restrict__ tokens,
                            const int* __restrict__ fin_in, int* __restrict__ fin_out,
                            const int* __restrict__ tr_in, int* __restrict__ tr_out, int t) {
  const int i = threadIdx.x;
  if (i >= ROWS) return;
  const int b = i / CK;
  const int src = b * CK + parent[i];
  const int tok = tokens[i];
  fin_out[i] = fin_in[src] | (tok == 2 ? 1 : 0);
  for (int tt = 0; tt < CT; ++tt)
    tr_out[i * CT + tt] = (tt == t) ? tok : tr_in[src * CT + tt];
}

__global__ void init_kernel(float* h1c, float* c1c, float* h2c, float* c2c,
                            int* tokens, float* logp, int* finA, int* trA) {
  const int idx = blockIdx.x * blockDim.x + threadIdx.x;
  if (idx < ROWS * C_H) { h1c[idx] = 0.f; c1c[idx] = 0.f; h2c[idx] = 0.f; c2c[idx] = 0.f; }
  if (idx < ROWS) { tokens[idx] = 1; logp[idx] = 0.f; finA[idx] = 0; }
  if (idx < ROWS * CT) trA[idx] = 0;
}

__global__ void vmean_kernel(const float* __restrict__ features, float* __restrict__ vmean) {
  const int idx = blockIdx.x * blockDim.x + threadIdx.x;
  if (idx >= CB * C_VD) return;
  const int b = idx >> 11, d = idx & 2047;
  float s = 0.0f;
  for (int r = 0; r < CR; ++r) s += features[((size_t)b * CR + r) * C_VD + d];
  vmean[idx] = s * (1.0f / (float)CR);
}

__global__ void finalize_kernel(const int* __restrict__ trajs, const float* __restrict__ logp,
                                float* __restrict__ out) {
  const int idx = blockIdx.x * blockDim.x + threadIdx.x;
  if (idx >= CB * CT * CK + ROWS) return;
  if (idx < CB * CT * CK) {
    const int b = idx / (CT * CK);
    const int rem = idx % (CT * CK);
    const int tt = rem / CK, k = rem % CK;
    out[idx] = (float)trajs[(b * CK + k) * CT + tt];
  } else {
    out[idx] = logp[idx - CB * CT * CK];
  }
}

// ---------------------------------------------------------------------------
extern "C" void kernel_launch(void* const* d_in, const int* in_sizes, int n_in,
                              void* d_out, int out_size, void* d_ws, size_t ws_size,
                              hipStream_t stream) {
  (void)in_sizes; (void)n_in; (void)out_size;
  const float* features = (const float*)d_in[0];
  const float* E    = (const float*)d_in[1];
  const float* Wi1  = (const float*)d_in[2];
  const float* Wh1  = (const float*)d_in[3];
  const float* b1   = (const float*)d_in[4];
  const float* Wi2  = (const float*)d_in[5];
  const float* Wh2  = (const float*)d_in[6];
  const float* b2   = (const float*)d_in[7];
  const float* Wva  = (const float*)d_in[8];
  const float* bva  = (const float*)d_in[9];
  const float* Wha  = (const float*)d_in[10];
  const float* bha  = (const float*)d_in[11];
  const float* wa   = (const float*)d_in[12];
  const float* Wout = (const float*)d_in[13];
  const float* bout = (const float*)d_in[14];
  float* out = (float*)d_out;

  char* wsb = (char*)d_ws;
  size_t off = 0;
  auto alloc = [&](size_t bytes) -> void* {
    off = (off + 255) & ~(size_t)255;
    void* p = wsb + off;
    off += bytes;
    return p;
  };

  // --- always-present buffers (~22 MB) ---
  bf16* xcat1p = (bf16*)alloc((size_t)6 * 144 * 32 * 16 * 2);
  bf16* xcat2p = (bf16*)alloc((size_t)6 * 128 * 32 * 16 * 2);
  bf16* h1p    = (bf16*)alloc((size_t)6 * 32 * 32 * 16 * 2);
  bf16* h2p    = (bf16*)alloc((size_t)6 * 32 * 32 * 16 * 2);
  bf16* featsP = (bf16*)alloc((size_t)72 * 64 * 32 * 16 * 2);
  float* featP = (float*)alloc((size_t)CB * CR * C_PH * 4);
  float* vmean = (float*)alloc((size_t)CB * C_VD * 4);
  float* gates = (float*)alloc((size_t)ROWS * 4096 * 4);
  float* hp    = (float*)alloc((size_t)ROWS * C_PH * 4);
  float* vhat  = (float*)alloc((size_t)ROWS * C_VD * 4);
  float* logits = (float*)alloc((size_t)ROWS * CV * 4);
  float* scores = (float*)alloc((size_t)ROWS * 10016 * 4);
  float* h1c = (float*)alloc((size_t)ROWS * C_H * 4);
  float* h1t = (float*)alloc((size_t)ROWS * C_H * 4);
  float* c1c = (float*)alloc((size_t)ROWS * C_H * 4);
  float* c1t = (float*)alloc((size_t)ROWS * C_H * 4);
  float* h2c = (float*)alloc((size_t)ROWS * C_H * 4);
  float* h2t = (float*)alloc((size_t)ROWS * C_H * 4);
  float* c2c = (float*)alloc((size_t)ROWS * C_H * 4);
  float* c2t = (float*)alloc((size_t)ROWS * C_H * 4);
  int* tokens  = (int*)alloc(ROWS * 4);
  int* parent  = (int*)alloc(ROWS * 4);
  float* logp  = (float*)alloc(ROWS * 4);
  int* finA = (int*)alloc(ROWS * 4);
  int* finB = (int*)alloc(ROWS * 4);
  int* trA = (int*)alloc((size_t)ROWS * CT * 4);
  int* trB = (int*)alloc((size_t)ROWS * CT * 4);

  // --- packed bf16 weights (~93 MB), used only if workspace allows ---
  const int NT1 = 256, KT1 = 144;   // [Wi1|Wh1]  N=4096, K=4608
  const int NT2 = 256, KT2 = 128;   // [Wi2|Wh2]  N=4096, K=4096
  const int NTO = 626, KTO = 32;    // Wout       N=10016(pad), K=1024
  const int NTH = 16,  KTH = 32;    // Wha        N=256,  K=1024
  const int NTV = 16,  KTV = 64;    // Wva        N=256,  K=2048
  bf16* Wc1p = (bf16*)alloc((size_t)NT1 * KT1 * 512 * 2);
  bf16* Wc2p = (bf16*)alloc((size_t)NT2 * KT2 * 512 * 2);
  bf16* Wotp = (bf16*)alloc((size_t)NTO * KTO * 512 * 2);
  bf16* Whap = (bf16*)alloc((size_t)NTH * KTH * 512 * 2);
  bf16* Wvap = (bf16*)alloc((size_t)NTV * KTV * 512 * 2);
  const bool packed = (off <= ws_size);

  // ---- launch-invariant prep ----
  init_kernel<<<(ROWS * C_H + 255) / 256, 256, 0, stream>>>(h1c, c1c, h2c, c2c,
                                                            tokens, logp, finA, trA);
  vmean_kernel<<<(CB * C_VD + 255) / 256, 256, 0, stream>>>(features, vmean);
  pack_a_kernel<<<(72 * 64 * 32 + 255) / 256, 256, 0, stream>>>(features, featsP, 72, 64, C_VD);

  if (packed) {
    pack_b_kernel<<<(NT1 * KT1 * 32 + 255) / 256, 256, 0, stream>>>(Wi1, 3584, Wh1, 1024, Wc1p, NT1, KT1, 4096);
    pack_b_kernel<<<(NT2 * KT2 * 32 + 255) / 256, 256, 0, stream>>>(Wi2, 3072, Wh2, 1024, Wc2p, NT2, KT2, 4096);
    pack_b_kernel<<<(NTO * KTO * 32 + 255) / 256, 256, 0, stream>>>(Wout, 1024, nullptr, 0, Wotp, NTO, KTO, CV);
    pack_b_kernel<<<(NTH * KTH * 32 + 255) / 256, 256, 0, stream>>>(Wha, 1024, nullptr, 0, Whap, NTH, KTH, C_PH);
    pack_b_kernel<<<(NTV * KTV * 32 + 255) / 256, 256, 0, stream>>>(Wva, 2048, nullptr, 0, Wvap, NTV, KTV, C_PH);
    gemm_pk_wmma<6><<<dim3(NTV, 12), 32, 0, stream>>>(featsP, Wvap, KTV, bva, featP, C_PH);
  } else {
    gemm_bf16_wmma<6><<<dim3(16, 12), 32, 0, stream>>>(featsP, 64, Wva, C_VD, nullptr, 0, bva, featP, C_PH);
  }

  int* fin_cur = finA; int* fin_alt = finB;
  int* tr_cur = trA;  int* tr_alt = trB;

  for (int t = 0; t < CT; ++t) {
    // LSTM1: gates = [h2, vmean, emb, h1] @ [Wi1 | Wh1]^T + b1
    concat1_pack<<<(6 * 144 * 32 + 255) / 256, 256, 0, stream>>>(tokens, h1c, h2c, vmean, E, xcat1p);
    if (packed)
      gemm_pk_wmma<6><<<dim3(NT1, 1), 32, 0, stream>>>(xcat1p, Wc1p, KT1, b1, gates, 4096);
    else
      gemm_bf16_wmma<6><<<dim3(256, 1), 32, 0, stream>>>(xcat1p, 144, Wi1, 3584, Wh1, 1024, b1, gates, 4096);
    lstm_ew<<<(ROWS * C_H + 255) / 256, 256, 0, stream>>>(gates, c1c, h1t, c1t);

    // attention: hp = h1 @ Wha.T + bha ; alpha/vhat
    pack_a_kernel<<<(6 * 32 * 32 + 255) / 256, 256, 0, stream>>>(h1t, h1p, 6, 32, C_H);
    if (packed)
      gemm_pk_wmma<6><<<dim3(NTH, 1), 32, 0, stream>>>(h1p, Whap, KTH, bha, hp, C_PH);
    else
      gemm_bf16_wmma<6><<<dim3(16, 1), 32, 0, stream>>>(h1p, 32, Wha, C_H, nullptr, 0, bha, hp, C_PH);
    attention_kernel<<<ROWS, 256, 0, stream>>>(featP, hp, wa, features, vhat);

    // LSTM2: gates = [h1, vhat, h2prev] @ [Wi2 | Wh2]^T + b2
    concat2_pack<<<(6 * 128 * 32 + 255) / 256, 256, 0, stream>>>(h1t, vhat, h2c, xcat2p);
    if (packed)
      gemm_pk_wmma<6><<<dim3(NT2, 1), 32, 0, stream>>>(xcat2p, Wc2p, KT2, b2, gates, 4096);
    else
      gemm_bf16_wmma<6><<<dim3(256, 1), 32, 0, stream>>>(xcat2p, 128, Wi2, 3072, Wh2, 1024, b2, gates, 4096);
    lstm_ew<<<(ROWS * C_H + 255) / 256, 256, 0, stream>>>(gates, c2c, h2t, c2t);

    // logits = h2 @ Wout.T + bout
    pack_a_kernel<<<(6 * 32 * 32 + 255) / 256, 256, 0, stream>>>(h2t, h2p, 6, 32, C_H);
    if (packed)
      gemm_pk_wmma<6><<<dim3(NTO, 1), 32, 0, stream>>>(h2p, Wotp, KTO, bout, logits, CV);
    else
      gemm_bf16_wmma<6><<<dim3(626, 1), 32, 0, stream>>>(h2p, 32, Wout, C_H, nullptr, 0, bout, logits, CV);

    // scoring + beam step
    score_kernel<<<ROWS, 256, 0, stream>>>(logits, tokens, fin_cur, logp, t, scores);
    topk_kernel<<<CB, 256, 0, stream>>>(scores, tokens, logp, parent);
    gather_states<<<(ROWS * C_H + 255) / 256, 256, 0, stream>>>(
        parent, h1t, h1c, c1t, c1c, h2t, h2c, c2t, c2c);
    beam_update<<<1, 128, 0, stream>>>(parent, tokens, fin_cur, fin_alt, tr_cur, tr_alt, t);

    int* tswap = fin_cur; fin_cur = fin_alt; fin_alt = tswap;
    tswap = tr_cur; tr_cur = tr_alt; tr_alt = tswap;
  }

  finalize_kernel<<<(CB * CT * CK + ROWS + 255) / 256, 256, 0, stream>>>(tr_cur, logp, out);
}
